// ResidualAttentionblk_52321291599986
// MI455X (gfx1250) — matrix operand order repaired and checked by reading, not compiled, not run on hardware
//
#include <hip/hip_runtime.h>

// ---------- types ----------
typedef __bf16 bf16_t;
typedef __attribute__((ext_vector_type(16))) __bf16 v16bf;
typedef __attribute__((ext_vector_type(8)))  float  v8f;
typedef __attribute__((ext_vector_type(4)))  unsigned int u32x4;
typedef __attribute__((ext_vector_type(4)))  float  f32x4;

#define NPATCH 676
#define KSEL   540
#define NLOW   136
#define SSEQ   542
#define BSZ    32
#define CH     1024
#define NHEAD  16
#define HD     64
#define T_ATT  (SSEQ*BSZ)         // 17344 tokens through attention
#define T_ALL  ((NPATCH+1)*BSZ)   // 21664 tokens through MLP
#define LN_EPS 1e-5f

// ---------- WMMA fragment loaders (bf16, K=32 step) ----------
// A (16x32): lane r=lane&15 is row M=r; half hf=lane>>4 holds K = hf*8+{0..7} and 16+hf*8+{0..7}
__device__ __forceinline__ v16bf ld_fragA(const bf16_t* rowptr, int hf) {
  union { v16bf v; u32x4 q[2]; } u;
  u.q[0] = *(const u32x4*)(rowptr + hf * 8);
  u.q[1] = *(const u32x4*)(rowptr + 16 + hf * 8);
  return u.v;
}
// B (32x16): lane n=lane&15 is col N=n; half hf holds K = hf*8+{0..7} and 16+hf*8+{0..7}
__device__ __forceinline__ v16bf ld_fragB(const bf16_t* rowptr, int hf) {
  union { v16bf v; u32x4 q[2]; } u;
  u.q[0] = *(const u32x4*)(rowptr + hf * 8);
  u.q[1] = *(const u32x4*)(rowptr + 16 + hf * 8);
  return u.v;
}
#define WMMA_BF16(a, b, c) \
  __builtin_amdgcn_wmma_f32_16x16x32_bf16(false, (a), false, (b), (short)0, (c), false, false)

// ---------- small kernels ----------
__global__ void k_cvt_bf16(const float* __restrict__ a, bf16_t* __restrict__ o, int n) {
  int i = blockIdx.x * 256 + threadIdx.x;
  if (i < n) o[i] = (bf16_t)a[i];
}

__global__ void k_copy_f32(const float* __restrict__ a, float* __restrict__ o, int n4) {
  int i = blockIdx.x * 256 + threadIdx.x;
  if (i < n4) ((f32x4*)o)[i] = ((const f32x4*)a)[i];
}

// mean over L=77 lang tokens, includes 1/77
__global__ void k_langmean(const float* __restrict__ lang, float* __restrict__ lm) {
  int b = blockIdx.x;
  for (int c = threadIdx.x; c < CH; c += 256) {
    float s = 0.f;
    for (int l = 0; l < 77; l++) s += lang[(size_t)l * (BSZ * CH) + b * CH + c];
    lm[b * CH + c] = s * (1.0f / 77.0f);
  }
}

// score[n,b] = xt[n,b,:] . langmean[b,:]  (one wave per dot)
__global__ void k_score(const float* __restrict__ x, const float* __restrict__ lm,
                        float* __restrict__ score) {
  int tid = threadIdx.x, lane = tid & 31, wid = tid >> 5;
  int g = blockIdx.x * 8 + wid;
  if (g >= NPATCH * BSZ) return;
  int n = g >> 5, b = g & 31;
  const float* xr = x + (size_t)(1 + n) * (BSZ * CH) + b * CH;
  const float* lr = lm + b * CH;
  float s = 0.f;
  for (int c = lane; c < CH; c += 32) s += xr[c] * lr[c];
  #pragma unroll
  for (int m = 1; m < 32; m <<= 1) s += __shfl_xor(s, m, 32);
  if (lane == 0) score[n * BSZ + b] = s;
}

// per-batch stable descending rank -> high/low index lists (ascending) + low softmax weights
__global__ void k_select(const float* __restrict__ score, int* __restrict__ high,
                         int* __restrict__ low, float* __restrict__ low_w) {
  int b = blockIdx.x, tid = threadIdx.x;
  __shared__ float s[NPATCH];
  __shared__ int   rk[NPATCH];
  __shared__ int   pos[NPATCH];
  __shared__ float ex[NPATCH];
  __shared__ float red[2];
  for (int n = tid; n < NPATCH; n += 256) s[n] = score[n * BSZ + b];
  __syncthreads();
  for (int n = tid; n < NPATCH; n += 256) {
    float sn = s[n]; int r = 0;
    for (int m = 0; m < NPATCH; m++) {
      float sm = s[m];
      r += (sm > sn) || (sm == sn && m < n);
    }
    rk[n] = r;
  }
  __syncthreads();
  for (int n = tid; n < NPATCH; n += 256) {
    int cnt = 0;
    for (int m = 0; m < n; m++) cnt += (rk[m] < KSEL);
    if (rk[n] < KSEL) { high[b * KSEL + cnt] = n; pos[n] = cnt; }
    else              { low[b * NLOW + (n - cnt)] = n; pos[n] = n - cnt; }
    ex[n] = 0.f;
  }
  __syncthreads();
  if (tid == 0) {
    float mx = -1e30f;
    for (int n = 0; n < NPATCH; n++) if (rk[n] >= KSEL) mx = fmaxf(mx, s[n]);
    red[0] = mx;
  }
  __syncthreads();
  float mx = red[0];
  for (int n = tid; n < NPATCH; n += 256) if (rk[n] >= KSEL) ex[n] = __expf(s[n] - mx);
  __syncthreads();
  if (tid == 0) {
    float sum = 0.f;
    for (int n = 0; n < NPATCH; n++) sum += ex[n];
    red[1] = sum;
  }
  __syncthreads();
  float inv = 1.0f / red[1];
  for (int n = tid; n < NPATCH; n += 256)
    if (rk[n] >= KSEL) low_w[b * NLOW + pos[n]] = ex[n] * inv;
}

// build seq (cls | gathered high | pooled low), LayerNorm, emit bf16 rows [s*B+b][CH]
__global__ __launch_bounds__(256) void k_build_seq(const float* __restrict__ x,
    const int* __restrict__ high, const int* __restrict__ low, const float* __restrict__ low_w,
    const float* __restrict__ lnw, const float* __restrict__ lnb, bf16_t* __restrict__ out) {
  int s = blockIdx.x, b = blockIdx.y, tid = threadIdx.x;
  float v[4];
  if (s == 0) {
    const float* r = x + (size_t)b * CH;
    #pragma unroll
    for (int i = 0; i < 4; i++) v[i] = r[tid + 256 * i];
  } else if (s <= KSEL) {
    int n = high[b * KSEL + (s - 1)];
    const float* r = x + (size_t)(1 + n) * (BSZ * CH) + b * CH;
    #pragma unroll
    for (int i = 0; i < 4; i++) v[i] = r[tid + 256 * i];
  } else {
    #pragma unroll
    for (int i = 0; i < 4; i++) v[i] = 0.f;
    for (int j = 0; j < NLOW; j++) {
      int n = low[b * NLOW + j];
      float w = low_w[b * NLOW + j];
      const float* r = x + (size_t)(1 + n) * (BSZ * CH) + b * CH;
      #pragma unroll
      for (int i = 0; i < 4; i++) v[i] += w * r[tid + 256 * i];
    }
  }
  __shared__ float rs[256], rq[256];
  float ps = 0.f, pq = 0.f;
  #pragma unroll
  for (int i = 0; i < 4; i++) { ps += v[i]; pq += v[i] * v[i]; }
  rs[tid] = ps; rq[tid] = pq;
  __syncthreads();
  for (int off = 128; off; off >>= 1) {
    if (tid < off) { rs[tid] += rs[tid + off]; rq[tid] += rq[tid + off]; }
    __syncthreads();
  }
  float mean = rs[0] * (1.0f / CH);
  float var  = rq[0] * (1.0f / CH) - mean * mean;
  float rinv = rsqrtf(var + LN_EPS);
  bf16_t* orow = out + ((size_t)s * BSZ + b) * CH;
  #pragma unroll
  for (int i = 0; i < 4; i++) {
    int c = tid + 256 * i;
    orow[c] = (bf16_t)((v[i] - mean) * rinv * lnw[c] + lnb[c]);
  }
}

// LayerNorm of f32 rows -> bf16 rows (for MLP input)
__global__ __launch_bounds__(256) void k_ln(const float* __restrict__ in,
    const float* __restrict__ lnw, const float* __restrict__ lnb, bf16_t* __restrict__ out) {
  int p = blockIdx.x, b = blockIdx.y, tid = threadIdx.x;
  const float* r = in + ((size_t)p * BSZ + b) * CH;
  float v[4];
  #pragma unroll
  for (int i = 0; i < 4; i++) v[i] = r[tid + 256 * i];
  __shared__ float rs[256], rq[256];
  float ps = 0.f, pq = 0.f;
  #pragma unroll
  for (int i = 0; i < 4; i++) { ps += v[i]; pq += v[i] * v[i]; }
  rs[tid] = ps; rq[tid] = pq;
  __syncthreads();
  for (int off = 128; off; off >>= 1) {
    if (tid < off) { rs[tid] += rs[tid + off]; rq[tid] += rq[tid + off]; }
    __syncthreads();
  }
  float mean = rs[0] * (1.0f / CH);
  float var  = rq[0] * (1.0f / CH) - mean * mean;
  float rinv = rsqrtf(var + LN_EPS);
  bf16_t* orow = out + ((size_t)p * BSZ + b) * CH;
  #pragma unroll
  for (int i = 0; i < 4; i++) {
    int c = tid + 256 * i;
    orow[c] = (bf16_t)((v[i] - mean) * rinv * lnw[c] + lnb[c]);
  }
}

// scatter attention output back with residual: out holds x already (copied)
__global__ void k_scatter(const float* __restrict__ d, const int* __restrict__ high,
                          const int* __restrict__ low, float* __restrict__ out) {
  int t = blockIdx.x, b = blockIdx.y, tid = threadIdx.x;
  int src, dstp;
  if (t == 0)            { src = 0;      dstp = 0; }
  else if (t <= KSEL)    { src = t;      dstp = 1 + high[b * KSEL + (t - 1)]; }
  else                   { src = SSEQ-1; dstp = 1 + low[b * NLOW + (t - 1 - KSEL)]; }
  const float* dr = d + ((size_t)src * BSZ + b) * CH;
  float* o = out + ((size_t)dstp * BSZ + b) * CH;
  #pragma unroll
  for (int i = 0; i < 4; i++) { int c = tid + 256 * i; o[c] += dr[c]; }
}

// ---------- generic WMMA GEMM: Y[M,N] = A[M,K](bf16) @ W[N,K]^T(bf16) + bias ----------
// block tile 128x128, 8 waves (4x2), wave tile 32x64, K-step 64, reg->LDS pipelined.
// MODE 0: store bf16; 1: store bf16 with x*sigmoid(1.702x); 2: store f32; 3: f32 +=
__device__ __forceinline__ void ld_tiles(const bf16_t* __restrict__ A,
                                         const bf16_t* __restrict__ W,
                                         int M, int K, int m0, int n0, int k0, int tid,
                                         u32x4 areg[4], u32x4 wreg[4]) {
  #pragma unroll
  for (int i = 0; i < 4; i++) {
    int c = tid + i * 256;
    int row = c >> 3, col = (c & 7) * 8;       // 128 rows x 8 chunks of 8 bf16
    int gr = m0 + row;
    if (gr >= M) gr = M - 1;                   // clamp: branch-free, epilogue masks stores
    areg[i] = *(const u32x4*)(A + (size_t)gr * K + k0 + col);
    wreg[i] = *(const u32x4*)(W + (size_t)(n0 + row) * K + k0 + col);
  }
}

template <int MODE>
__global__ __launch_bounds__(256) void k_gemm(const bf16_t* __restrict__ A,
    const bf16_t* __restrict__ W, const float* __restrict__ bias, void* __restrict__ out,
    int M, int N, int K) {
  __shared__ __align__(16) bf16_t As[128][72];
  __shared__ __align__(16) bf16_t Ws[128][72];
  int tid = threadIdx.x, lane = tid & 31, wid = tid >> 5;
  int wr = wid >> 1, wc = wid & 1;             // 4x2 wave grid, wave tile 32(M) x 64(N)
  int m0 = blockIdx.y * 128, n0 = blockIdx.x * 128;
  int r = lane & 15, hf = lane >> 4;

  v8f acc[2][4] = {};
  u32x4 areg[4], wreg[4];
  ld_tiles(A, W, M, K, m0, n0, 0, tid, areg, wreg);

  for (int k0 = 0; k0 < K; k0 += 64) {
    __syncthreads();
    #pragma unroll
    for (int i = 0; i < 4; i++) {
      int c = tid + i * 256;
      int row = c >> 3, col = (c & 7) * 8;
      *(u32x4*)&As[row][col] = areg[i];
      *(u32x4*)&Ws[row][col] = wreg[i];
    }
    __syncthreads();
    if (k0 + 64 < K) ld_tiles(A, W, M, K, m0, n0, k0 + 64, tid, areg, wreg);
    #pragma unroll
    for (int f = 0; f < 2; f++) {
      v16bf af0 = ld_fragA(&As[wr * 32 + r][f * 32], hf);
      v16bf af1 = ld_fragA(&As[wr * 32 + 16 + r][f * 32], hf);
      v16bf bfr[4];
      #pragma unroll
      for (int n = 0; n < 4; n++) bfr[n] = ld_fragB(&Ws[wc * 64 + n * 16 + r][f * 32], hf);
      #pragma unroll
      for (int n = 0; n < 4; n++) {
        acc[0][n] = WMMA_BF16(af0, bfr[n], acc[0][n]);
        acc[1][n] = WMMA_BF16(af1, bfr[n], acc[1][n]);
      }
    }
  }
  #pragma unroll
  for (int i = 0; i < 2; i++)
    #pragma unroll
    for (int n = 0; n < 4; n++) {
      int gn = n0 + wc * 64 + n * 16 + (lane & 15);
      float bv = bias[gn];
      #pragma unroll
      for (int v = 0; v < 8; v++) {
        int gm = m0 + wr * 32 + i * 16 + 8 * hf + v;   // C layout: M = vgpr + 8*(lane>=16)
        if (gm >= M) continue;
        float val = acc[i][n][v] + bv;
        size_t off = (size_t)gm * N + gn;
        if (MODE == 0)      ((bf16_t*)out)[off] = (bf16_t)val;
        else if (MODE == 1) ((bf16_t*)out)[off] = (bf16_t)(val * (1.0f / (1.0f + __expf(-1.702f * val))));
        else if (MODE == 2) ((float*)out)[off] = val;
        else                ((float*)out)[off] += val;
      }
    }
}

// ---------- flash attention: grid (qchunk=5, head=16, batch=32), 8 waves, 128 q-rows ----------
// K-tile = 64 keys: 18 WMMAs per barrier pair; row-sums via WMMA against all-ones B fragment.
__global__ __launch_bounds__(256) void k_attn(const bf16_t* __restrict__ qkv, bf16_t* __restrict__ O) {
  int qc = blockIdx.x, h = blockIdx.y, b = blockIdx.z;
  __shared__ __align__(16) bf16_t Ks[64][72];      // [key][d]
  __shared__ __align__(16) bf16_t Vt[64][72];      // [d][key]
  __shared__ __align__(16) bf16_t Ps[8][16][72];   // per-wave P staging (16 q x 64 keys)
  int tid = threadIdx.x, lane = tid & 31, wid = tid >> 5;
  int r = lane & 15, hf = lane >> 4, c15 = lane & 15;

  int srow = qc * 128 + wid * 16 + r;
  int sq = srow < SSEQ ? srow : SSEQ - 1;
  const bf16_t* qrow = qkv + ((size_t)sq * BSZ + b) * (3 * CH) + h * HD;
  v16bf qf0 = ld_fragA(qrow, hf);
  v16bf qf1 = ld_fragA(qrow + 32, hf);

  v16bf onesf;
  #pragma unroll
  for (int i = 0; i < 16; i++) onesf[i] = (bf16_t)1.0f;

  float rm[8], rl[8];
  #pragma unroll
  for (int v = 0; v < 8; v++) { rm[v] = -1e30f; rl[v] = 0.f; }
  v8f oacc[4] = {};

  for (int kt = 0; kt < 9; kt++) {                 // ceil(542/64) tiles
    int key0 = kt * 64;
    __syncthreads();
    {                                              // stage K tile 64x64 (clamped; masked later)
      #pragma unroll
      for (int i = 0; i < 2; i++) {
        int c = tid + i * 256;
        int kk = c >> 3, col = (c & 7) * 8;
        int key = key0 + kk; if (key >= SSEQ) key = SSEQ - 1;
        *(u32x4*)&Ks[kk][col] =
            *(const u32x4*)(qkv + ((size_t)key * BSZ + b) * (3 * CH) + CH + h * HD + col);
      }
    }
    {                                              // stage V tile transposed 64x64 (clamped)
      #pragma unroll
      for (int i = 0; i < 2; i++) {
        int c = tid + i * 256;
        int kk = c >> 3, dch = (c & 7) * 8;
        int key = key0 + kk; if (key >= SSEQ) key = SSEQ - 1;
        const bf16_t* vp = qkv + ((size_t)key * BSZ + b) * (3 * CH) + 2 * CH + h * HD + dch;
        #pragma unroll
        for (int u = 0; u < 8; u++) Vt[dch + u][kk] = vp[u];
      }
    }
    __syncthreads();
    // scores: 16 q-rows x 64 keys, inner dim D=64 -> 2 k-steps, 4 col-groups
    v8f sf[4] = {};
    #pragma unroll
    for (int f = 0; f < 2; f++) {
      v16bf qa = f ? qf1 : qf0;
      #pragma unroll
      for (int j = 0; j < 4; j++) {
        v16bf bk = ld_fragB(&Ks[j * 16 + c15][f * 32], hf);
        sf[j] = WMMA_BF16(qa, bk, sf[j]);
      }
    }
    #pragma unroll
    for (int j = 0; j < 4; j++) {
      int key = key0 + j * 16 + c15;
      #pragma unroll
      for (int v = 0; v < 8; v++)
        sf[j][v] = (key < SSEQ) ? sf[j][v] * 0.125f : -1e30f;
    }
    float al[8];
    #pragma unroll
    for (int v = 0; v < 8; v++) {
      float mx = fmaxf(fmaxf(sf[0][v], sf[1][v]), fmaxf(sf[2][v], sf[3][v]));
      #pragma unroll
      for (int m = 1; m < 16; m <<= 1) mx = fmaxf(mx, __shfl_xor(mx, m, 32));
      float mnew = fmaxf(rm[v], mx);
      al[v] = __expf(rm[v] - mnew);
      rm[v] = mnew;
      #pragma unroll
      for (int j = 0; j < 4; j++) {
        float p = __expf(sf[j][v] - mnew);
        Ps[wid][v + 8 * hf][j * 16 + c15] = (bf16_t)p;
      }
    }
    #pragma unroll
    for (int j2 = 0; j2 < 4; j2++)
      #pragma unroll
      for (int v = 0; v < 8; v++) oacc[j2][v] *= al[v];
    __syncthreads();
    v16bf pf0 = ld_fragA(&Ps[wid][r][0], hf);
    v16bf pf1 = ld_fragA(&Ps[wid][r][32], hf);
    // row sums via WMMA against all-ones B: C[row, n] = sum_k P[row,k], replicated over n
    v8f rsum = {};
    rsum = WMMA_BF16(pf0, onesf, rsum);
    rsum = WMMA_BF16(pf1, onesf, rsum);
    #pragma unroll
    for (int v = 0; v < 8; v++) rl[v] = rl[v] * al[v] + rsum[v];
    #pragma unroll
    for (int j2 = 0; j2 < 4; j2++) {
      v16bf bv0 = ld_fragB(&Vt[j2 * 16 + c15][0], hf);
      v16bf bv1 = ld_fragB(&Vt[j2 * 16 + c15][32], hf);
      oacc[j2] = WMMA_BF16(pf0, bv0, oacc[j2]);
      oacc[j2] = WMMA_BF16(pf1, bv1, oacc[j2]);
    }
  }
  float rinv[8];
  #pragma unroll
  for (int v = 0; v < 8; v++) rinv[v] = 1.0f / rl[v];
  #pragma unroll
  for (int j2 = 0; j2 < 4; j2++)
    #pragma unroll
    for (int v = 0; v < 8; v++) {
      int row = qc * 128 + wid * 16 + v + 8 * hf;
      if (row < SSEQ) {
        float o = oacc[j2][v] * rinv[v];
        O[((size_t)row * BSZ + b) * CH + h * HD + j2 * 16 + c15] = (bf16_t)o;
      }
    }
}

// ---------- host launcher ----------
extern "C" void kernel_launch(void* const* d_in, const int* in_sizes, int n_in,
                              void* d_out, int out_size, void* d_ws, size_t ws_size,
                              hipStream_t stream) {
  const float* x     = (const float*)d_in[0];
  const float* lang  = (const float*)d_in[1];
  const float* ln1w  = (const float*)d_in[2];
  const float* ln1b  = (const float*)d_in[3];
  const float* ln2w  = (const float*)d_in[4];
  const float* ln2b  = (const float*)d_in[5];
  const float* wqkv  = (const float*)d_in[6];
  const float* bqkv  = (const float*)d_in[7];
  const float* wout  = (const float*)d_in[8];
  const float* bout  = (const float*)d_in[9];
  const float* wfc1  = (const float*)d_in[10];
  const float* bfc1  = (const float*)d_in[11];
  const float* wfc2  = (const float*)d_in[12];
  const float* bfc2  = (const float*)d_in[13];
  float* out = (float*)d_out;

  char* base = (char*)d_ws;
  size_t off = 0;
  auto alloc = [&](size_t bytes) { size_t cur = off; off += (bytes + 255) & ~(size_t)255; return cur; };
  float*  lm     = (float*)(base + alloc((size_t)BSZ * CH * 4));
  float*  score  = (float*)(base + alloc((size_t)NPATCH * BSZ * 4));
  int*    high   = (int*)  (base + alloc((size_t)BSZ * KSEL * 4));
  int*    low    = (int*)  (base + alloc((size_t)BSZ * NLOW * 4));
  float*  low_w  = (float*)(base + alloc((size_t)BSZ * NLOW * 4));
  bf16_t* wqkv_b = (bf16_t*)(base + alloc((size_t)3 * CH * CH * 2));
  bf16_t* wout_b = (bf16_t*)(base + alloc((size_t)CH * CH * 2));
  bf16_t* wfc1_b = (bf16_t*)(base + alloc((size_t)4 * CH * CH * 2));
  bf16_t* wfc2_b = (bf16_t*)(base + alloc((size_t)4 * CH * CH * 2));
  char* regA = base + alloc((size_t)T_ATT * CH * 2);                 // seq_ln, then attn O
  char* regC = base + alloc((size_t)T_ALL * CH * 2);                 // h_ln
  char* regB = base + alloc((size_t)T_ALL * 4 * CH * 2);             // qkv -> d(f32) -> h1act

  // weight conversion to bf16
  k_cvt_bf16<<<(3 * CH * CH + 255) / 256, 256, 0, stream>>>(wqkv, wqkv_b, 3 * CH * CH);
  k_cvt_bf16<<<(CH * CH + 255) / 256, 256, 0, stream>>>(wout, wout_b, CH * CH);
  k_cvt_bf16<<<(4 * CH * CH + 255) / 256, 256, 0, stream>>>(wfc1, wfc1_b, 4 * CH * CH);
  k_cvt_bf16<<<(4 * CH * CH + 255) / 256, 256, 0, stream>>>(wfc2, wfc2_b, 4 * CH * CH);

  // token scoring + selection
  k_langmean<<<BSZ, 256, 0, stream>>>(lang, lm);
  k_score<<<(NPATCH * BSZ) / 8, 256, 0, stream>>>(x, lm, score);
  k_select<<<BSZ, 256, 0, stream>>>(score, high, low, low_w);

  // seq build + LN1 -> bf16
  bf16_t* seqln = (bf16_t*)regA;
  k_build_seq<<<dim3(SSEQ, BSZ), 256, 0, stream>>>(x, high, low, low_w, ln1w, ln1b, seqln);

  // QKV GEMM: [T_ATT,1024] x [3072,1024]^T -> bf16 qkv
  bf16_t* qkvb = (bf16_t*)regB;
  k_gemm<0><<<dim3(3 * CH / 128, (T_ATT + 127) / 128), 256, 0, stream>>>(
      seqln, wqkv_b, bqkv, qkvb, T_ATT, 3 * CH, CH);

  // attention (flash, WMMA) -> bf16 O in regA
  bf16_t* Ob = (bf16_t*)regA;
  k_attn<<<dim3((SSEQ + 127) / 128, NHEAD, BSZ), 256, 0, stream>>>(qkvb, Ob);

  // output projection -> f32 d in regB
  float* dbuf = (float*)regB;
  k_gemm<2><<<dim3(CH / 128, (T_ATT + 127) / 128), 256, 0, stream>>>(
      Ob, wout_b, bout, dbuf, T_ATT, CH, CH);

  // residual: out = x, then scatter-add d
  k_copy_f32<<<(T_ALL * CH / 4 + 255) / 256, 256, 0, stream>>>(x, out, T_ALL * CH / 4);
  k_scatter<<<dim3(NPATCH + 1, BSZ), 256, 0, stream>>>(dbuf, high, low, out);

  // LN2 -> bf16 h_ln
  bf16_t* hln = (bf16_t*)regC;
  k_ln<<<dim3(NPATCH + 1, BSZ), 256, 0, stream>>>(out, ln2w, ln2b, hln);

  // MLP fc1 with gelu-sigmoid -> bf16 h1act (regB, d is dead)
  bf16_t* h1act = (bf16_t*)regB;
  k_gemm<1><<<dim3(4 * CH / 128, (T_ALL + 127) / 128), 256, 0, stream>>>(
      hln, wfc1_b, bfc1, h1act, T_ALL, 4 * CH, CH);

  // MLP fc2, accumulate into out (residual)
  k_gemm<3><<<dim3(CH / 128, (T_ALL + 127) / 128), 256, 0, stream>>>(
      h1act, wfc2_b, bfc2, out, T_ALL, CH, 4 * CH);
}
